// MultiheadAttention_69836168233168
// MI455X (gfx1250) — matrix operand order repaired and checked by reading, not compiled, not run on hardware
//
#include <hip/hip_runtime.h>

// MultiheadAttention  B=2 S=2048 E=1024 H=16 D=64  (fp32 in/out)
// bf16 WMMA (v_wmma_f32_16x16x32_bf16) everywhere; GEMMs use double-buffered
// LDS staging filled with global_load_async_to_lds_b128 (ASYNCcnt pipeline),
// last K-step peeled so the steady-state loop is branch-free.
// Workspace: 3*8M (Xq/Xk/Xv) + 4*2M (weights) + 3*8M (Qp/Kp/Vt) + 8M (Attn) = 64 MB.

typedef __attribute__((ext_vector_type(16))) __bf16 v16bf;
typedef __attribute__((ext_vector_type(8)))  __bf16 v8bf;
typedef __attribute__((ext_vector_type(4)))  __bf16 v4bf;
typedef __attribute__((ext_vector_type(8)))  float  v8f;

static __device__ __forceinline__ __bf16 f2bf(float f) {
  unsigned u = __builtin_bit_cast(unsigned, f);
  u += 0x7FFFu + ((u >> 16) & 1u);              // round-to-nearest-even
  unsigned short h = (unsigned short)(u >> 16);
  return __builtin_bit_cast(__bf16, h);
}

static __device__ __forceinline__ v16bf combine16(v8bf lo, v8bf hi) {
  return __builtin_shufflevector(lo, hi, 0,1,2,3,4,5,6,7,8,9,10,11,12,13,14,15);
}

// LDS byte offset of a generic pointer to __shared__ (aperture rule: addr[31:0])
static __device__ __forceinline__ unsigned ldsoff(const void* p) {
  return (unsigned)(unsigned long long)p;
}

// async 16B copy global -> LDS (GV addressing), tracked by ASYNCcnt
static __device__ __forceinline__ void async_cp16(unsigned lds_addr, const void* gptr) {
  asm volatile("global_load_async_to_lds_b128 %0, %1, off"
               :: "v"(lds_addr), "v"(gptr) : "memory");
}

// ---------------------------------------------------------------- convert ---
__global__ __launch_bounds__(256)
void cvt_f32_bf16(const float* __restrict__ in, __bf16* __restrict__ out, int n) {
  int i = (blockIdx.x * 256 + threadIdx.x) * 4;
  if (i >= n) return;
  float4 f = *(const float4*)(in + i);
  v4bf o;
  o[0] = f2bf(f.x); o[1] = f2bf(f.y); o[2] = f2bf(f.z); o[3] = f2bf(f.w);
  *(v4bf*)(out + i) = o;
}

// ------------------------------------------------------------------- GEMM ---
// y[m,n] = sum_k A[m,k]*W[n,k]   A:4096x1024  W:1024x1024 (bf16 row-major)
// Block 256 thr = 8 waves (2x4), block tile 128x256, wave tile 64x64, K-step 32.
// A/B K-step tiles staged in LDS via async copies, double buffered.
// mode 1: bf16 Qp/Kp[b,h,s,d]   mode 2: bf16 Vt[b,h,d,s]   mode 3: f32 [4096,1024]
#define LDSP 40                                  // bf16 pitch (80 B) -> bank-spread

__global__ __launch_bounds__(256)
void gemm_bf16_wmma(const __bf16* __restrict__ A, const __bf16* __restrict__ W,
                    __bf16* __restrict__ outb, float* __restrict__ outf, int mode) {
  __shared__ __bf16 ldsA[2][128 * LDSP];         // 2 x 10 KB
  __shared__ __bf16 ldsB[2][256 * LDSP];         // 2 x 20 KB
  const int K    = 1024;
  const int tid  = threadIdx.x;
  const int lane = tid & 31;
  const int wid  = tid >> 5;
  const int lr   = lane & 15;
  const int hlf  = lane >> 4;
  const int wm   = (wid >> 2) * 64;              // wave row offset in block tile
  const int wn   = (wid & 3) * 64;               // wave col offset in block tile
  const int m0   = blockIdx.y * 128;
  const int n0   = blockIdx.x * 256;

  // copy assignment: A tile 128 rows x 64B: thread t -> row t>>1, 32B half (t&1)
  //                  B tile 256 rows x 64B: thread t -> whole row t
  const int arow = tid >> 1;
  const int acol = (tid & 1) * 16;               // element offset

  auto issue_tile = [&](int kb, int buf) {
    const __bf16* ga = A + (size_t)(m0 + arow) * K + kb + acol;
    unsigned la = ldsoff(&ldsA[buf][arow * LDSP + acol]);
    async_cp16(la,      ga);
    async_cp16(la + 16, ga + 8);
    const __bf16* gb = W + (size_t)(n0 + tid) * K + kb;
    unsigned lb = ldsoff(&ldsB[buf][tid * LDSP]);
    async_cp16(lb,      gb);
    async_cp16(lb + 16, gb + 8);
    async_cp16(lb + 32, gb + 16);
    async_cp16(lb + 48, gb + 24);
  };                                             // 6 async ops / thread / tile

  v8f acc[4][4];
  #pragma unroll
  for (int i = 0; i < 4; ++i)
    #pragma unroll
    for (int j = 0; j < 4; ++j) acc[i][j] = {};

  auto compute_tile = [&](int buf) {
    v16bf a[4], b[4];
    #pragma unroll
    for (int i = 0; i < 4; ++i) {                // A frag rows wm+i*16+lr
      const __bf16* p = &ldsA[buf][(wm + i * 16 + lr) * LDSP];
      v8bf lo = *(const v8bf*)(p + 16 * hlf);
      v8bf hi = *(const v8bf*)(p + 16 * hlf + 8);
      a[i] = combine16(lo, hi);
    }
    #pragma unroll
    for (int j = 0; j < 4; ++j) {                // B frag rows wn+j*16+lr
      const __bf16* p = &ldsB[buf][(wn + j * 16 + lr) * LDSP + 16 * hlf];
      b[j] = combine16(*(const v8bf*)(p), *(const v8bf*)(p + 8));
    }
    #pragma unroll
    for (int i = 0; i < 4; ++i)
      #pragma unroll
      for (int j = 0; j < 4; ++j)
        acc[i][j] = __builtin_amdgcn_wmma_f32_16x16x32_bf16(
            false, a[i], false, b[j], (short)0, acc[i][j], false, false);
  };

  issue_tile(0, 0);
  // steady state: branch-free body, next tile always in flight
  for (int kb = 0; kb < K - 32; kb += 32) {
    const int buf = (kb >> 5) & 1;
    issue_tile(kb + 32, buf ^ 1);
    asm volatile("s_wait_asynccnt 0x6" ::: "memory");   // tile kb resident
    __syncthreads();
    compute_tile(buf);
    __syncthreads();                             // buf reusable for tile kb+2
  }
  // drain: last tile (index 31 -> buffer 1)
  asm volatile("s_wait_asynccnt 0x0" ::: "memory");
  __syncthreads();
  compute_tile(1);

  #pragma unroll
  for (int i = 0; i < 4; ++i)
    #pragma unroll
    for (int j = 0; j < 4; ++j)
      #pragma unroll
      for (int r = 0; r < 8; ++r) {
        const int m = m0 + wm + i * 16 + r + 8 * hlf;
        const int n = n0 + wn + j * 16 + lr;
        const float v = acc[i][j][r];
        if (mode == 1) {                         // Qp/Kp [b,h,s,d]
          int bb = m >> 11, s = m & 2047, h = n >> 6, d = n & 63;
          outb[(((size_t)bb * 16 + h) * 2048 + s) * 64 + d] = f2bf(v);
        } else if (mode == 2) {                  // Vt [b,h,d,s]
          int bb = m >> 11, s = m & 2047, h = n >> 6, d = n & 63;
          outb[(((size_t)bb * 16 + h) * 64 + d) * 2048 + s] = f2bf(v);
        } else {                                 // f32 row-major
          outf[(size_t)m * 1024 + n] = v;
        }
      }
}

// -------------------------------------------------------- flash attention ---
// Block 128 thr = 4 waves; each wave owns 32 queries of one (b,h).
// Per 64-key tile: 16 WMMA QK^T, online softmax, P->LDS->A-frag, 16 WMMA PV.
__global__ __launch_bounds__(128)
void attn_flash_wmma(const __bf16* __restrict__ Qp, const __bf16* __restrict__ Kp,
                     const __bf16* __restrict__ Vt, __bf16* __restrict__ Attn) {
  __shared__ __bf16 smem[4 * 32 * 72];           // 32x64 P tile per wave, pitch 72
  const int lane = threadIdx.x & 31;
  const int wid  = threadIdx.x >> 5;
  const int lr   = lane & 15;
  const int hlf  = lane >> 4;
  const int bh   = blockIdx.y;                   // 0..31
  const int bb   = bh >> 4, h = bh & 15;
  const int q0   = blockIdx.x * 128 + wid * 32;

  const __bf16* Qh = Qp + (size_t)bh * 2048 * 64;
  const __bf16* Kh = Kp + (size_t)bh * 2048 * 64;
  const __bf16* Vh = Vt + (size_t)bh * 64 * 2048;
  __bf16* lds = smem + wid * 32 * 72;

  v8f o[2][4];
  float mrow[2][8], lrow[2][8];
  #pragma unroll
  for (int i = 0; i < 2; ++i) {
    #pragma unroll
    for (int j = 0; j < 4; ++j) o[i][j] = {};
    #pragma unroll
    for (int r = 0; r < 8; ++r) { mrow[i][r] = -1e30f; lrow[i][r] = 0.0f; }
  }

  v16bf aq[2][2];                                // Q frags, reused all key tiles
  #pragma unroll
  for (int i = 0; i < 2; ++i)
    #pragma unroll
    for (int ks = 0; ks < 2; ++ks) {
      const __bf16* p = Qh + (size_t)(q0 + i * 16 + lr) * 64 + ks * 32;
      v8bf lo = *(const v8bf*)(p + 16 * hlf);
      v8bf hi = *(const v8bf*)(p + 16 * hlf + 8);
      aq[i][ks] = combine16(lo, hi);
    }

  const float c = 0.125f * 1.4426950408889634f;  // (1/sqrt(D)) * log2(e)

  for (int kt = 0; kt < 32; ++kt) {
    const int k0 = kt * 64;

    v8f s[2][4];
    #pragma unroll
    for (int i = 0; i < 2; ++i)
      #pragma unroll
      for (int j = 0; j < 4; ++j) s[i][j] = {};
    #pragma unroll
    for (int ks = 0; ks < 2; ++ks) {
      v16bf bk[4];
      #pragma unroll
      for (int j = 0; j < 4; ++j)
        bk[j] = *(const v16bf*)(Kh + (size_t)(k0 + j * 16 + lr) * 64 + ks * 32 + 16 * hlf);
      #pragma unroll
      for (int i = 0; i < 2; ++i)
        #pragma unroll
        for (int j = 0; j < 4; ++j)
          s[i][j] = __builtin_amdgcn_wmma_f32_16x16x32_bf16(
              false, aq[i][ks], false, bk[j], (short)0, s[i][j], false, false);
    }

    // online softmax: row r+8*hlf of tile i, reduce across 16 lanes
    #pragma unroll
    for (int i = 0; i < 2; ++i)
      #pragma unroll
      for (int r = 0; r < 8; ++r) {
        float tmax = fmaxf(fmaxf(s[i][0][r], s[i][1][r]),
                           fmaxf(s[i][2][r], s[i][3][r])) * c;
        #pragma unroll
        for (int off = 1; off < 16; off <<= 1)
          tmax = fmaxf(tmax, __shfl_xor(tmax, off, 32));
        const float mnew  = fmaxf(mrow[i][r], tmax);
        const float alpha = exp2f(mrow[i][r] - mnew);
        mrow[i][r] = mnew;
        float rs = 0.0f;
        #pragma unroll
        for (int j = 0; j < 4; ++j) {
          float pv = exp2f(s[i][j][r] * c - mnew);
          s[i][j][r] = pv;
          rs += pv;
        }
        #pragma unroll
        for (int off = 1; off < 16; off <<= 1)
          rs += __shfl_xor(rs, off, 32);
        lrow[i][r] = lrow[i][r] * alpha + rs;
        #pragma unroll
        for (int j = 0; j < 4; ++j) o[i][j][r] *= alpha;
      }

    // P (C layout) -> LDS -> A layout (same-wave DS ops are in-order)
    #pragma unroll
    for (int i = 0; i < 2; ++i)
      #pragma unroll
      for (int j = 0; j < 4; ++j)
        #pragma unroll
        for (int r = 0; r < 8; ++r)
          lds[(i * 16 + r + 8 * hlf) * 72 + j * 16 + lr] = f2bf(s[i][j][r]);

    #pragma unroll
    for (int ks = 0; ks < 2; ++ks) {
      v16bf ap[2];
      #pragma unroll
      for (int i = 0; i < 2; ++i) {
        const __bf16* p = lds + (size_t)(i * 16 + lr) * 72 + ks * 32;
        v8bf lo = *(const v8bf*)(p + 16 * hlf);
        v8bf hi = *(const v8bf*)(p + 16 * hlf + 8);
        ap[i] = combine16(lo, hi);
      }
      v16bf bv[4];
      #pragma unroll
      for (int j = 0; j < 4; ++j)                // Vt rows contiguous in k
        bv[j] = *(const v16bf*)(Vh + (size_t)(j * 16 + lr) * 2048 + k0 + ks * 32 + 16 * hlf);
      #pragma unroll
      for (int i = 0; i < 2; ++i)
        #pragma unroll
        for (int j = 0; j < 4; ++j)
          o[i][j] = __builtin_amdgcn_wmma_f32_16x16x32_bf16(
              false, ap[i], false, bv[j], (short)0, o[i][j], false, false);
    }
  }

  #pragma unroll
  for (int i = 0; i < 2; ++i)
    #pragma unroll
    for (int j = 0; j < 4; ++j)
      #pragma unroll
      for (int r = 0; r < 8; ++r) {
        const int srow = q0 + i * 16 + r + 8 * hlf;
        const int d    = j * 16 + lr;
        const float v  = o[i][j][r] / lrow[i][r];
        Attn[((size_t)bb * 2048 + srow) * 1024 + h * 64 + d] = f2bf(v);
      }
}

// ----------------------------------------------------------------- launch ---
extern "C" void kernel_launch(void* const* d_in, const int* in_sizes, int n_in,
                              void* d_out, int out_size, void* d_ws, size_t ws_size,
                              hipStream_t stream) {
  const float* q  = (const float*)d_in[0];
  const float* k  = (const float*)d_in[1];
  const float* v  = (const float*)d_in[2];
  const float* Wq = (const float*)d_in[3];
  const float* Wk = (const float*)d_in[4];
  const float* Wv = (const float*)d_in[5];
  const float* Wo = (const float*)d_in[6];

  const int NX = 4096 * 1024;
  const int NW = 1024 * 1024;

  __bf16* Xq  = (__bf16*)d_ws;
  __bf16* Xk  = Xq  + NX;
  __bf16* Xv  = Xk  + NX;
  __bf16* Wqb = Xv  + NX;
  __bf16* Wkb = Wqb + NW;
  __bf16* Wvb = Wkb + NW;
  __bf16* Wob = Wvb + NW;
  __bf16* Qp  = Wob + NW;
  __bf16* Kp  = Qp  + NX;
  __bf16* Vt  = Kp  + NX;
  __bf16* At  = Vt  + NX;       // total 64 MB of workspace

  cvt_f32_bf16<<<NX / 1024, 256, 0, stream>>>(q,  Xq,  NX);
  cvt_f32_bf16<<<NX / 1024, 256, 0, stream>>>(k,  Xk,  NX);
  cvt_f32_bf16<<<NX / 1024, 256, 0, stream>>>(v,  Xv,  NX);
  cvt_f32_bf16<<<NW / 1024, 256, 0, stream>>>(Wq, Wqb, NW);
  cvt_f32_bf16<<<NW / 1024, 256, 0, stream>>>(Wk, Wkb, NW);
  cvt_f32_bf16<<<NW / 1024, 256, 0, stream>>>(Wv, Wvb, NW);
  cvt_f32_bf16<<<NW / 1024, 256, 0, stream>>>(Wo, Wob, NW);

  dim3 gg(1024 / 256, 4096 / 128);
  gemm_bf16_wmma<<<gg, 256, 0, stream>>>(Xq, Wqb, Qp, nullptr, 1);
  gemm_bf16_wmma<<<gg, 256, 0, stream>>>(Xk, Wkb, Kp, nullptr, 1);
  gemm_bf16_wmma<<<gg, 256, 0, stream>>>(Xv, Wvb, Vt, nullptr, 2);

  attn_flash_wmma<<<dim3(2048 / 128, 32), 128, 0, stream>>>(Qp, Kp, Vt, At);

  gemm_bf16_wmma<<<gg, 256, 0, stream>>>(At, Wob, nullptr, (float*)d_out, 3);
}
